// SparseGroupMHA_38886633898146
// MI455X (gfx1250) — matrix operand order repaired and checked by reading, not compiled, not run on hardware
//
#include <hip/hip_runtime.h>
#include <hip/hip_bf16.h>
#include <math.h>

typedef __bf16 bf16_t;
typedef __attribute__((ext_vector_type(16))) __bf16 v16bf;
typedef __attribute__((ext_vector_type(8)))  float  v8f;

static __device__ __forceinline__ v8f wmma_bf16(v16bf a, v16bf b, v8f c) {
    return __builtin_amdgcn_wmma_f32_16x16x32_bf16(false, a, false, b, (short)0, c, false, false);
}

// Wave-local LDS fence: DS ops of a wave complete in-order; waiting DScnt==0
// makes this wave's LDS stores visible to its own cross-lane loads without a
// block barrier (needed inside wave-divergent-trip-count loops).
static __device__ __forceinline__ void wave_lds_fence() {
    asm volatile("s_wait_dscnt 0x0" ::: "memory");
}

// ---------------------------------------------------------------------------
// fp32 -> bf16 bulk convert
// ---------------------------------------------------------------------------
__global__ __launch_bounds__(256) void cvt_f32_bf16(const float* __restrict__ src,
                                                    bf16_t* __restrict__ dst, int n4) {
    int i = (blockIdx.x * 256 + threadIdx.x);
    if (i < n4) {
        const float4 v = ((const float4*)src)[i];
        bf16_t* d = dst + (size_t)i * 4;
        d[0] = (bf16_t)v.x; d[1] = (bf16_t)v.y; d[2] = (bf16_t)v.z; d[3] = (bf16_t)v.w;
    }
}

// ---------------------------------------------------------------------------
// W[K][N] (fp32) -> Wt[N][K] (bf16) 32x32 LDS tile transpose
// ---------------------------------------------------------------------------
__global__ __launch_bounds__(256) void transpose_cvt(const float* __restrict__ W,
                                                     bf16_t* __restrict__ Wt, int K, int N) {
    __shared__ bf16_t tile[32][33];
    const int bx = blockIdx.x % (N / 32);
    const int by = blockIdx.x / (N / 32);
    const int tx = threadIdx.x & 31;
    const int ty = threadIdx.x >> 5;
    #pragma unroll
    for (int i = 0; i < 32; i += 8)
        tile[ty + i][tx] = (bf16_t)W[(size_t)(by * 32 + ty + i) * N + bx * 32 + tx];
    __syncthreads();
    #pragma unroll
    for (int i = 0; i < 32; i += 8)
        Wt[(size_t)(bx * 32 + ty + i) * K + by * 32 + tx] = tile[tx][ty + i];
}

// ---------------------------------------------------------------------------
// GEMM (TN): C[M,N] = A[M,K] * Bt[N,K]^T, bf16 in, K-contiguous both sides
// -> all fragments are b128 loads.  Wave: 16x64 strip; block: 128x64.
// ---------------------------------------------------------------------------
template <typename TC>
__global__ __launch_bounds__(256) void gemm_bf16_tn(
        const bf16_t* __restrict__ A, const bf16_t* __restrict__ Bt,
        TC* __restrict__ C, int M, int N, int K) {
    const int lane = threadIdx.x & 31;
    const int wave = threadIdx.x >> 5;
    const int half = lane >> 4;
    const int lm   = lane & 15;

    const int mblocks = M >> 7;
    const int msuper  = blockIdx.x % mblocks;
    const int ntile   = blockIdx.x / mblocks;
    const int row0    = (msuper * 8 + wave) * 16;
    const int col0    = ntile * 64;

    v8f acc0 = {}, acc1 = {}, acc2 = {}, acc3 = {};

    for (int kk = 0; kk < K; kk += 32) {
        v16bf a;
        const bf16_t* arow = A + (size_t)(row0 + lm) * K + kk;
        #pragma unroll
        for (int j = 0; j < 8; ++j) {
            const int k0 = 16 * half + 2 * j;
            a[2 * j]     = arow[k0];
            a[2 * j + 1] = arow[k0 + 1];
        }
        #pragma unroll
        for (int sub = 0; sub < 4; ++sub) {
            v16bf b;
            const bf16_t* brow = Bt + (size_t)(col0 + sub * 16 + lm) * K + kk;
            #pragma unroll
            for (int j = 0; j < 8; ++j) {
                const int k0 = 16 * half + 2 * j;
                b[2 * j]     = brow[k0];
                b[2 * j + 1] = brow[k0 + 1];
            }
            v8f& acc = sub == 0 ? acc0 : sub == 1 ? acc1 : sub == 2 ? acc2 : acc3;
            acc = wmma_bf16(a, b, acc);
        }
    }

    #pragma unroll
    for (int sub = 0; sub < 4; ++sub) {
        const v8f& acc = sub == 0 ? acc0 : sub == 1 ? acc1 : sub == 2 ? acc2 : acc3;
        #pragma unroll
        for (int r = 0; r < 8; ++r)
            C[(size_t)(row0 + 8 * half + r) * N + col0 + sub * 16 + lm] = (TC)acc[r];
    }
}

// ---------------------------------------------------------------------------
// Block-sparse group attention.  group_ids is SORTED, so each row attends to
// one contiguous [lo, hi) column range (mean width B/NUM_GROUPS = 16).  Per
// 16-row tile we binary-search the range and process only 32-column chunks
// intersecting it (online softmax, flash rescaling).  All range math is
// wave-uniform -> EXEC all-ones around every WMMA; no block barriers inside
// the dynamic loop (wave-local s_wait_dscnt instead).
// ---------------------------------------------------------------------------
#define VT_PAD 264   // 132 dwords/row -> 16 lanes land 4 banks apart
#define AP_PAD 40    //  20 dwords/row -> 4-bank spacing, conflict-free b128

__global__ __launch_bounds__(128) void attn_group_kernel(
        const bf16_t* __restrict__ Qb, const bf16_t* __restrict__ Kb,
        const bf16_t* __restrict__ Vb, const int* __restrict__ gids_g,
        bf16_t* __restrict__ Ob) {
    __shared__ bf16_t VtS[64][VT_PAD];      // V^T : [dk][b]  (~33 KB)
    __shared__ bf16_t attnS[4][16][AP_PAD]; // per-wave 16x32 P chunk (~5 KB)
    __shared__ int    gids[256];

    const int t    = blockIdx.x >> 3;       // H = 8
    const int h    = blockIdx.x & 7;
    const int lane = threadIdx.x & 31;
    const int wave = threadIdx.x >> 5;
    const int half = lane >> 4;
    const int lm   = lane & 15;

    const size_t baseTH   = (size_t)t * 256 * 512 + (size_t)h * 64;
    const float  sm_scale = 0.125f;

    // stage V(t,h) transposed into LDS: VtS[dk][b]
    for (int i = threadIdx.x; i < 256 * 16; i += 128) {
        const int b  = i >> 4;
        const int d4 = (i & 15) * 4;
        const bf16_t* src = Vb + baseTH + (size_t)b * 512 + d4;
        VtS[d4 + 0][b] = src[0];
        VtS[d4 + 1][b] = src[1];
        VtS[d4 + 2][b] = src[2];
        VtS[d4 + 3][b] = src[3];
    }
    for (int i = threadIdx.x; i < 256; i += 128) gids[i] = gids_g[i];
    __syncthreads();

    for (int rt = 0; rt < 4; ++rt) {
        const int m0 = (wave * 4 + rt) * 16;

        // ---- allowed column range for this row tile (wave-uniform) ----
        const int g_lo = gids[m0];
        const int g_hi = gids[m0 + 15];
        int lo, hi;
        {   // lower_bound(g_lo)
            int L = 0, R = 256;
            #pragma unroll
            for (int it = 0; it < 8; ++it) {
                const int mid = (L + R) >> 1;
                if (gids[mid] < g_lo) L = mid + 1; else R = mid;
            }
            lo = L;
        }
        {   // upper_bound(g_hi)
            int L = 0, R = 256;
            #pragma unroll
            for (int it = 0; it < 8; ++it) {
                const int mid = (L + R) >> 1;
                if (gids[mid] <= g_hi) L = mid + 1; else R = mid;
            }
            hi = L;
        }
        const int c_beg = (lo >> 4) & ~1;        // 32-aligned chunk of 16-col tiles
        const int c_end = (hi + 15) >> 4;

        // ---- Q fragments for this row tile (reused across all chunks) ----
        v16bf qa0, qa1;
        {
            const bf16_t* qrow = Qb + baseTH + (size_t)(m0 + lm) * 512;
            #pragma unroll
            for (int j = 0; j < 8; ++j) {
                const int k0 = 16 * half + 2 * j;
                qa0[2 * j]     = qrow[k0];
                qa0[2 * j + 1] = qrow[k0 + 1];
                qa1[2 * j]     = qrow[32 + k0];
                qa1[2 * j + 1] = qrow[32 + k0 + 1];
            }
        }

        float m_r[8], s_r[8];
        #pragma unroll
        for (int r = 0; r < 8; ++r) { m_r[r] = -3.0e38f; s_r[r] = 0.f; }
        v8f o0 = {}, o1 = {}, o2 = {}, o3 = {};

        for (int c = c_beg; c < c_end; c += 2) {     // 32 columns per chunk
            // ---- scores for column tiles c and c+1 ----
            v8f sa = {}, sb = {};
            #pragma unroll
            for (int kt = 0; kt < 2; ++kt) {
                const int kk = kt * 32;
                v16bf ba, bb;   // K^T columns = K rows: contiguous loads
                const bf16_t* krow0 = Kb + baseTH + (size_t)(c * 16 + lm) * 512 + kk;
                const bf16_t* krow1 = krow0 + (size_t)16 * 512;
                #pragma unroll
                for (int j = 0; j < 8; ++j) {
                    const int k0 = 16 * half + 2 * j;
                    ba[2 * j]     = krow0[k0];
                    ba[2 * j + 1] = krow0[k0 + 1];
                    bb[2 * j]     = krow1[k0];
                    bb[2 * j + 1] = krow1[k0 + 1];
                }
                sa = wmma_bf16(kt == 0 ? qa0 : qa1, ba, sa);
                sb = wmma_bf16(kt == 0 ? qa0 : qa1, bb, sb);
            }

            // ---- mask + online softmax over these 32 columns ----
            const int ga = gids[c * 16 + lm];
            const int gb = gids[c * 16 + 16 + lm];
            #pragma unroll
            for (int r = 0; r < 8; ++r) {
                const int gr = gids[m0 + 8 * half + r];
                float va = sa[r] * sm_scale;
                float vb = sb[r] * sm_scale;
                va = (ga == gr) ? va : -1.0e30f;
                vb = (gb == gr) ? vb : -1.0e30f;
                float pm = fmaxf(va, vb);
                #pragma unroll
                for (int off = 1; off < 16; off <<= 1)   // within 16-lane half
                    pm = fmaxf(pm, __shfl_xor(pm, off, 32));
                const float nm    = fmaxf(m_r[r], pm);
                const float alpha = __expf(m_r[r] - nm);
                m_r[r] = nm;
                const float pa = (va > -1.0e29f) ? __expf(va - nm) : 0.f;
                const float pb = (vb > -1.0e29f) ? __expf(vb - nm) : 0.f;
                attnS[wave][8 * half + r][lm]      = (bf16_t)pa;
                attnS[wave][8 * half + r][16 + lm] = (bf16_t)pb;
                float ps = pa + pb;
                #pragma unroll
                for (int off = 1; off < 16; off <<= 1)
                    ps += __shfl_xor(ps, off, 32);
                s_r[r] = s_r[r] * alpha + ps;
                o0[r] *= alpha; o1[r] *= alpha; o2[r] *= alpha; o3[r] *= alpha;
            }
            wave_lds_fence();   // this wave's P writes -> this wave's reads

            // ---- O += P(16x32) @ V[c*16 : c*16+32, :] ----
            v16bf a;
            #pragma unroll
            for (int j = 0; j < 8; ++j) {
                const int k0 = 16 * half + 2 * j;
                a[2 * j]     = attnS[wave][lm][k0];
                a[2 * j + 1] = attnS[wave][lm][k0 + 1];
            }
            #pragma unroll
            for (int sub = 0; sub < 4; ++sub) {
                v16bf b;
                const int n = sub * 16 + lm;            // dk
                #pragma unroll
                for (int j = 0; j < 8; ++j) {
                    const int kv = c * 16 + 16 * half + 2 * j;
                    b[2 * j]     = VtS[n][kv];
                    b[2 * j + 1] = VtS[n][kv + 1];
                }
                v8f& o = sub == 0 ? o0 : sub == 1 ? o1 : sub == 2 ? o2 : o3;
                o = wmma_bf16(a, b, o);
            }
            wave_lds_fence();   // reads done before next chunk overwrites P
        }

        // ---- finalize: divide by running sum, store bf16 ----
        #pragma unroll
        for (int sub = 0; sub < 4; ++sub) {
            const v8f& o = sub == 0 ? o0 : sub == 1 ? o1 : sub == 2 ? o2 : o3;
            #pragma unroll
            for (int r = 0; r < 8; ++r) {
                const int row = m0 + 8 * half + r;
                Ob[baseTH + (size_t)row * 512 + sub * 16 + lm] =
                    (bf16_t)(o[r] / s_r[r]);
            }
        }
    }
}

// ---------------------------------------------------------------------------
extern "C" void kernel_launch(void* const* d_in, const int* in_sizes, int n_in,
                              void* d_out, int out_size, void* d_ws, size_t ws_size,
                              hipStream_t stream) {
    const float* X    = (const float*)d_in[0];   // [256,256,512]
    const int*   gids = (const int*)  d_in[1];   // [256]
    const float* Wq   = (const float*)d_in[2];   // [512,512]
    const float* Wk   = (const float*)d_in[3];
    const float* Wv   = (const float*)d_in[4];
    const float* Wo   = (const float*)d_in[5];
    float*       out  = (float*)d_out;

    const int M = 256 * 256, N = 512, K = 512;
    const size_t MN = (size_t)M * N;

    bf16_t* Xb  = (bf16_t*)d_ws;          // 64 MB, reused as Ab afterwards
    bf16_t* Qb  = Xb + MN;
    bf16_t* Kb  = Qb + MN;
    bf16_t* Vb  = Kb + MN;
    bf16_t* Wqt = Vb + MN;                // 4 x 512 KB transposed bf16 weights
    bf16_t* Wkt = Wqt + (size_t)N * K;
    bf16_t* Wvt = Wkt + (size_t)N * K;
    bf16_t* Wot = Wvt + (size_t)N * K;
    bf16_t* Ab  = Xb;                     // alias: X dead after QKV projections

    const int n4 = (int)(MN / 4);
    cvt_f32_bf16<<<(n4 + 255) / 256, 256, 0, stream>>>(X, Xb, n4);

    const int tgrid = (N / 32) * (K / 32);
    transpose_cvt<<<tgrid, 256, 0, stream>>>(Wq, Wqt, K, N);
    transpose_cvt<<<tgrid, 256, 0, stream>>>(Wk, Wkt, K, N);
    transpose_cvt<<<tgrid, 256, 0, stream>>>(Wv, Wvt, K, N);
    transpose_cvt<<<tgrid, 256, 0, stream>>>(Wo, Wot, K, N);

    const int ggrid = (M / 128) * (N / 64);
    gemm_bf16_tn<bf16_t><<<ggrid, 256, 0, stream>>>(Xb, Wqt, Qb, M, N, K);
    gemm_bf16_tn<bf16_t><<<ggrid, 256, 0, stream>>>(Xb, Wkt, Kb, M, N, K);
    gemm_bf16_tn<bf16_t><<<ggrid, 256, 0, stream>>>(Xb, Wvt, Vb, M, N, K);

    attn_group_kernel<<<256 * 8, 128, 0, stream>>>(Qb, Kb, Vb, gids, Ab);

    gemm_bf16_tn<float><<<ggrid, 256, 0, stream>>>(Ab, Wot, out, M, N, K);
}
